// PerformerAttention_47304769798228
// MI455X (gfx1250) — matrix operand order repaired and checked by reading, not compiled, not run on hardware
//
#include <hip/hip_runtime.h>

// Performer causal linear attention, CDNA5 (gfx1250), wave32 + WMMA f16->f32.
// All GEMMs (feature projection, Q@S, QK^T, K^T@V state updates) lowered to
// v_wmma_f32_16x16x32_f16. Causal scan parallelized via chunked prefix states.
// Q substep blocks are staged into LDS with double-buffered async copies
// (GLOBAL_LOAD_ASYNC_TO_LDS_B128 / ASYNCcnt) when the toolchain exposes them.

#define B_    4
#define T_    4096
#define H_    12
#define E_    64
#define MREAL 266
#define MP    288      // padded feature dim: 9 * 32
#define BH    48
#define CHUNK 128
#define NC    32       // T / CHUNK
#define DAUG  80       // 64 V cols + ones col (den) + zero pad to 5 tiles

#define NORMALIZER 0.35355339059327373f   // 64^-0.25
#define RATIO      0.06131393f            // 266^-0.5

typedef __attribute__((ext_vector_type(16))) _Float16 v16h;
typedef __attribute__((ext_vector_type(8)))  _Float16 v8h;
typedef __attribute__((ext_vector_type(8)))  float    v8f;
typedef int v4i_ __attribute__((__vector_size__(16)));   // matches builtin pointee

#if defined(__has_builtin)
#if __has_builtin(__builtin_amdgcn_global_load_async_to_lds_b128) && \
    __has_builtin(__builtin_amdgcn_s_wait_asynccnt)
#define HAVE_ASYNC_LDS 1
#endif
#endif
#ifndef HAVE_ASYNC_LDS
#define HAVE_ASYNC_LDS 0
#endif

__device__ __forceinline__ void async_copy_b128(_Float16* lds_dst,
                                                const _Float16* gsrc) {
#if HAVE_ASYNC_LDS
  __builtin_amdgcn_global_load_async_to_lds_b128(
      (__attribute__((address_space(1))) v4i_*)(void*)gsrc,
      (__attribute__((address_space(3))) v4i_*)(void*)lds_dst,
      0, 0);
#else
  *(v8h*)lds_dst = *(const v8h*)gsrc;   // synchronous fallback
#endif
}
__device__ __forceinline__ void async_join() {
#if HAVE_ASYNC_LDS
  __builtin_amdgcn_s_wait_asynccnt(0);
#endif
}

__device__ __forceinline__ v8f wmma16(v16h a, v16h b, v8f c) {
  return __builtin_amdgcn_wmma_f32_16x16x32_f16(false, a, false, b, (short)0, c,
                                                false, false);
}

// A fragment [16 x 32] f16 from row-major tile (ld in elements).
// lane<16: row=lane,   K = {0..7, 16..23};  lane>=16: row=lane-16, K={8..15,24..31}
__device__ __forceinline__ v16h load_fragA(const _Float16* tile, int ld) {
  const int lane = threadIdx.x & 31;
  const _Float16* p = tile + (size_t)(lane & 15) * ld + ((lane & 16) ? 8 : 0);
  v8h lo = *(const v8h*)(p);
  v8h hi = *(const v8h*)(p + 16);
  return __builtin_shufflevector(lo, hi, 0,1,2,3,4,5,6,7,8,9,10,11,12,13,14,15);
}

// B fragment [32 x 16] f16 given B^T stored row-major (column-contiguous in K).
// lane<16: col=lane, K = {0..7, 16..23};  lane>=16: col=lane-16, K={8..15,24..31}
__device__ __forceinline__ v16h load_fragB(const _Float16* tileT, int ld) {
  const int lane = threadIdx.x & 31;
  const _Float16* p = tileT + (size_t)(lane & 15) * ld + ((lane & 16) ? 8 : 0);
  v8h lo = *(const v8h*)(p);
  v8h hi = *(const v8h*)(p + 16);
  return __builtin_shufflevector(lo, hi, 0,1,2,3,4,5,6,7,8,9,10,11,12,13,14,15);
}

__device__ __forceinline__ unsigned enc_ord(float f) {
  unsigned u = __float_as_uint(f);
  return (u & 0x80000000u) ? ~u : (u | 0x80000000u);
}
__device__ __forceinline__ float dec_ord(unsigned u) {
  return (u & 0x80000000u) ? __uint_as_float(u & 0x7fffffffu)
                           : __uint_as_float(~u);
}

// ---------------- Phase 1a: global max of k's data_dash per (b,h) -----------
__global__ __launch_bounds__(256) void kmax_kernel(
    const float* __restrict__ key, const float* __restrict__ proj,
    unsigned* __restrict__ kmaxEnc) {
  __shared__ __align__(16) _Float16 pl[MP * E_];
  __shared__ __align__(16) _Float16 xd[CHUNK * E_];
  __shared__ float wred[8];
  const int tid = threadIdx.x;
  const int t0 = blockIdx.x * CHUNK;
  const int bh = blockIdx.y, b = bh / H_, h = bh % H_;
  for (int i = tid; i < MP * E_; i += 256) {
    int m = i >> 6, e = i & 63;
    pl[i] = (m < MREAL) ? (_Float16)proj[m * E_ + e] : (_Float16)0.f;
  }
  for (int i = tid; i < CHUNK * E_; i += 256) {
    int r = i >> 6, e = i & 63;
    xd[i] = (_Float16)(NORMALIZER * key[(((size_t)b * T_ + t0 + r) * H_ + h) * E_ + e]);
  }
  __syncthreads();
  const int w = tid >> 5, lane = tid & 31, n0 = lane & 15;
  float mx = -1e30f;
  for (int nt = 0; nt < 18; nt++) {
    v8f acc = {};
#pragma unroll
    for (int ks = 0; ks < 2; ks++) {
      v16h a  = load_fragA(xd + (w * 16) * E_ + ks * 32, E_);
      v16h bb = load_fragB(pl + (nt * 16) * E_ + ks * 32, E_);
      acc = wmma16(a, bb, acc);
    }
    if (nt * 16 + n0 < MREAL) {
#pragma unroll
      for (int v = 0; v < 8; v++) mx = fmaxf(mx, acc[v]);
    }
  }
#pragma unroll
  for (int off = 16; off >= 1; off >>= 1) mx = fmaxf(mx, __shfl_xor(mx, off, 32));
  if (lane == 0) wred[w] = mx;
  __syncthreads();
  if (tid == 0) {
    float m2 = wred[0];
    for (int i = 1; i < 8; i++) m2 = fmaxf(m2, wred[i]);
    atomicMax(&kmaxEnc[bh], enc_ord(m2));
  }
}

// ---------------- Phase 1b: softmax features (q or k) -----------------------
__global__ __launch_bounds__(256) void feat_kernel(
    const float* __restrict__ data, const float* __restrict__ proj,
    const unsigned* __restrict__ kmaxEnc, _Float16* __restrict__ Fout,
    _Float16* __restrict__ FoutT, int isQuery) {
  __shared__ __align__(16) _Float16 pl[MP * E_];
  __shared__ __align__(16) _Float16 xd[CHUNK * E_];
  __shared__ float diag[CHUNK];
  const int tid = threadIdx.x;
  const int t0 = blockIdx.x * CHUNK;
  const int bh = blockIdx.y, b = bh / H_, h = bh % H_;
  for (int i = tid; i < MP * E_; i += 256) {
    int m = i >> 6, e = i & 63;
    pl[i] = (m < MREAL) ? (_Float16)proj[m * E_ + e] : (_Float16)0.f;
  }
  for (int i = tid; i < CHUNK * E_; i += 256) {
    int r = i >> 6, e = i & 63;
    xd[i] = (_Float16)(NORMALIZER * data[(((size_t)b * T_ + t0 + r) * H_ + h) * E_ + e]);
  }
  __syncthreads();
  if (tid < CHUNK) {
    float s = 0.f;
#pragma unroll
    for (int e = 0; e < E_; e++) { float v = (float)xd[tid * E_ + e]; s += v * v; }
    diag[tid] = 0.5f * s;
  }
  __syncthreads();
  const int w = tid >> 5, lane = tid & 31;
  const int n0 = lane & 15, mb = (lane & 16) ? 8 : 0;
  v8f dd[18];
#pragma unroll
  for (int nt = 0; nt < 18; nt++) {
    v8f acc = {};
#pragma unroll
    for (int ks = 0; ks < 2; ks++) {
      v16h a  = load_fragA(xd + (w * 16) * E_ + ks * 32, E_);
      v16h bb = load_fragB(pl + (nt * 16) * E_ + ks * 32, E_);
      acc = wmma16(a, bb, acc);
    }
    dd[nt] = acc;
  }
  float rmx[8];
  if (isQuery) {
#pragma unroll
    for (int v = 0; v < 8; v++) rmx[v] = -1e30f;
#pragma unroll
    for (int nt = 0; nt < 18; nt++) {
      if (nt * 16 + n0 < MREAL) {
#pragma unroll
        for (int v = 0; v < 8; v++) rmx[v] = fmaxf(rmx[v], dd[nt][v]);
      }
    }
#pragma unroll
    for (int v = 0; v < 8; v++) {
#pragma unroll
      for (int off = 8; off >= 1; off >>= 1)
        rmx[v] = fmaxf(rmx[v], __shfl_xor(rmx[v], off, 32));
    }
  } else {
    float g = dec_ord(kmaxEnc[bh]);
#pragma unroll
    for (int v = 0; v < 8; v++) rmx[v] = g;
  }
#pragma unroll
  for (int nt = 0; nt < 18; nt++) {
    int n = nt * 16 + n0;
    v8h hv;
#pragma unroll
    for (int v = 0; v < 8; v++) {
      int row = w * 16 + mb + v;
      float f = (n < MREAL)
                    ? RATIO * (__expf(dd[nt][v] - diag[row] - rmx[v]) + 1e-4f)
                    : 0.f;
      _Float16 hf = (_Float16)f;
      Fout[((size_t)bh * T_ + t0 + row) * MP + n] = hf;
      hv[v] = hf;
    }
    if (FoutT)  // transposed copy (feature-major) for state-update B operands
      *(v8h*)(FoutT + ((size_t)bh * MP + n) * T_ + t0 + w * 16 + mb) = hv;
  }
}

// ---------------- Phase 2a: per-chunk partial states S^T = Vaug^T @ K -------
__global__ __launch_bounds__(320) void chunkstate_kernel(
    const float* __restrict__ value, const _Float16* __restrict__ KfT,
    float* __restrict__ Spart) {
  __shared__ __align__(16) _Float16 vT[DAUG * CHUNK];
  const int tid = threadIdx.x;
  const int c = blockIdx.x, bh = blockIdx.y, b = bh / H_, h = bh % H_;
  for (int i = tid; i < DAUG * CHUNK; i += 320) {
    int e = i >> 7, r = i & 127;
    float val = (e < E_) ? value[(((size_t)b * T_ + c * CHUNK + r) * H_ + h) * E_ + e]
                         : (e == E_ ? 1.f : 0.f);
    vT[i] = (_Float16)val;
  }
  __syncthreads();
  const int w = tid >> 5, lane = tid & 31;
  const int dt = w >> 1, mg = w & 1, n0 = lane & 15, mb = (lane & 16) ? 8 : 0;
  v8f z = {};
  v8f acc[9];
#pragma unroll
  for (int i = 0; i < 9; i++) acc[i] = z;
#pragma unroll
  for (int rs = 0; rs < 4; rs++) {
    v16h a = load_fragA(vT + (dt * 16) * CHUNK + rs * 32, CHUNK);
#pragma unroll
    for (int i = 0; i < 9; i++) {
      int mt = mg * 9 + i;
      v16h bb = load_fragB(KfT + ((size_t)bh * MP + mt * 16) * T_ + c * CHUNK + rs * 32, T_);
      acc[i] = wmma16(a, bb, acc[i]);
    }
  }
  float* outp = Spart + (size_t)(bh * NC + c) * DAUG * MP;
#pragma unroll
  for (int i = 0; i < 9; i++) {
    int mt = mg * 9 + i;
    float* p = outp + (size_t)(dt * 16) * MP + mt * 16 + n0;
#pragma unroll
    for (int v = 0; v < 8; v++) p[(size_t)(mb + v) * MP] = acc[i][v];
  }
}

// ---------------- Phase 2b: exclusive prefix over chunks --------------------
__global__ __launch_bounds__(256) void prefix_kernel(float* __restrict__ Spart) {
  const int bh = blockIdx.y;
  const int idx = blockIdx.x * 256 + threadIdx.x;  // < 80*288 = 23040
  size_t base = (size_t)bh * NC * DAUG * MP + idx;
  float run = 0.f;
  for (int c = 0; c < NC; c++) {
    float* p = Spart + base + (size_t)c * DAUG * MP;
    float v = *p;
    *p = run;
    run += v;
  }
}

// ---------------- Phase 3: chunked causal attention -------------------------
__global__ __launch_bounds__(320) void attn_kernel(
    const float* __restrict__ value, const _Float16* __restrict__ Qf,
    const _Float16* __restrict__ Kf, const _Float16* __restrict__ KfT,
    const float* __restrict__ Spre, float* __restrict__ out) {
  __shared__ __align__(16) _Float16 smir[DAUG * MP];    // S^T mirror, f16 (45KB)
  __shared__ __align__(16) _Float16 vT[DAUG * CHUNK];   // Vaug^T        (20KB)
  __shared__ __align__(16) _Float16 qs[2][32 * MP];     // Q double buf  (36KB)
  __shared__ __align__(16) _Float16 aqk[32 * 32];       // masked QK^T   ( 2KB)
  __shared__ float denl[32];
  const int tid = threadIdx.x, lane = tid & 31, w = tid >> 5;
  const int c = blockIdx.x, bh = blockIdx.y, b = bh / H_, h = bh % H_;
  const int n0 = lane & 15, mb = (lane & 16) ? 8 : 0;
  const int dt = w >> 1, mg = w & 1;   // state-update tile ownership (9 tiles/wave)
  const int oti = w & 1, odi = w >> 1; // output-tile role (2 t-tiles x 5 d-tiles)
  const float* pre = Spre + (size_t)(bh * NC + c) * DAUG * MP;
  const _Float16* Qb = Qf + ((size_t)bh * T_ + c * CHUNK) * MP;
  const _Float16* Kb = Kf + ((size_t)bh * T_ + c * CHUNK) * MP;
  // issue async staging of substep 0's Q block (32 x MP f16, b128 chunks)
  for (int i = tid; i < 32 * MP / 8; i += 320)
    async_copy_b128(&qs[0][i * 8], Qb + (size_t)i * 8);
  for (int i = tid; i < DAUG * MP; i += 320) smir[i] = (_Float16)pre[i];
  for (int i = tid; i < DAUG * CHUNK; i += 320) {
    int e = i >> 7, r = i & 127;
    float val = (e < E_) ? value[(((size_t)b * T_ + c * CHUNK + r) * H_ + h) * E_ + e]
                         : (e == E_ ? 1.f : 0.f);
    vT[i] = (_Float16)val;
  }
  v8f Sacc[9];
#pragma unroll
  for (int i = 0; i < 9; i++) {
    int mt = mg * 9 + i;
    const float* p = pre + (size_t)(dt * 16) * MP + mt * 16 + n0;
#pragma unroll
    for (int v = 0; v < 8; v++) Sacc[i][v] = p[(size_t)(mb + v) * MP];
  }
  for (int j = 0; j < 4; j++) {
    const int r0 = j * 32;
    // retire this substep's async Q copies; make LDS staging visible
    async_join();
    __syncthreads();
    // overlap: issue next substep's Q copy while this one computes
    if (j < 3) {
      const _Float16* qn = Qb + (size_t)(r0 + 32) * MP;
      for (int i = tid; i < 32 * MP / 8; i += 320)
        async_copy_b128(&qs[(j + 1) & 1][i * 8], qn + (size_t)i * 8);
    }
    const _Float16* qcur = &qs[j & 1][0];
    // (1) inter: OUT = Q_j @ S   (S^T f16 mirror holds all rows t < c*128+r0)
    v8f o = {};
#pragma unroll
    for (int ks = 0; ks < 9; ks++) {
      v16h a  = load_fragA(qcur + (size_t)(oti * 16) * MP + ks * 32, MP);
      v16h bb = load_fragB(smir + (odi * 16) * MP + ks * 32, MP);
      o = wmma16(a, bb, o);
    }
    // (2) intra: masked QK^T within the 32-row substep (waves 0,1 only)
    if (w < 2) {
      const int ti2 = w;
      for (int tj = 0; tj <= ti2; tj++) {
        v8f aq = {};
#pragma unroll
        for (int ks = 0; ks < 9; ks++) {
          v16h a  = load_fragA(qcur + (size_t)(ti2 * 16) * MP + ks * 32, MP);
          v16h bb = load_fragB(Kb + (size_t)(r0 + tj * 16) * MP + ks * 32, MP);
          aq = wmma16(a, bb, aq);
        }
#pragma unroll
        for (int v = 0; v < 8; v++) {
          int qi = ti2 * 16 + mb + v, kj = tj * 16 + n0;
          aqk[qi * 32 + kj] = (qi >= kj) ? (_Float16)aq[v] : (_Float16)0.f;
        }
      }
      if (ti2 == 0) {  // upper-right tile is fully masked
#pragma unroll
        for (int v = 0; v < 8; v++) aqk[(mb + v) * 32 + 16 + n0] = (_Float16)0.f;
      }
    }
    __syncthreads();
    // (3) OUT += Amask @ Vaug  (ones column of Vaug -> rowsum = intra denom)
    {
      v16h a  = load_fragA(aqk + oti * 16 * 32, 32);
      v16h bb = load_fragB(vT + (odi * 16) * CHUNK + r0, CHUNK);
      o = wmma16(a, bb, o);
    }
    // (4) denominator column (d==64 lives in odi==4, local n==0)
    if (odi == 4 && n0 == 0) {
#pragma unroll
      for (int v = 0; v < 8; v++) denl[oti * 16 + mb + v] = o[v] + 1e-6f;
    }
    __syncthreads();
    // (5) normalize + store out[b, t, h, e]
    if (odi < 4) {
#pragma unroll
      for (int v = 0; v < 8; v++) {
        int tr = r0 + oti * 16 + mb + v;
        float dv = denl[oti * 16 + mb + v];
        out[(((size_t)b * T_ + c * CHUNK + tr) * H_ + h) * E_ + odi * 16 + n0] =
            o[v] / dv;
      }
    }
    // (6) state update: S^T += Vaug_j^T @ K_j ; refresh f16 mirror
    // (safe: all waves finished reading smir for step (1) before the step-(2)
    //  barrier; next substep's reads are fenced by the top-of-loop barrier)
    if (j < 3) {
      v16h a = load_fragA(vT + (dt * 16) * CHUNK + r0, CHUNK);
#pragma unroll
      for (int i = 0; i < 9; i++) {
        int mt = mg * 9 + i;
        v16h bb = load_fragB(KfT + ((size_t)bh * MP + mt * 16) * T_ + c * CHUNK + r0, T_);
        Sacc[i] = wmma16(a, bb, Sacc[i]);
      }
#pragma unroll
      for (int i = 0; i < 9; i++) {
        int mt = mg * 9 + i;
#pragma unroll
        for (int v = 0; v < 8; v++)
          smir[(dt * 16 + mb + v) * MP + mt * 16 + n0] = (_Float16)Sacc[i][v];
      }
    }
  }
}

extern "C" void kernel_launch(void* const* d_in, const int* in_sizes, int n_in,
                              void* d_out, int out_size, void* d_ws, size_t ws_size,
                              hipStream_t stream) {
  const float* query = (const float*)d_in[0];
  const float* key   = (const float*)d_in[1];
  const float* value = (const float*)d_in[2];
  const float* proj  = (const float*)d_in[3];
  float* out = (float*)d_out;
  (void)in_sizes; (void)n_in; (void)out_size; (void)ws_size;

  char* ws = (char*)d_ws;
  unsigned* kmaxEnc = (unsigned*)ws;                      // 48 slots (+pad)
  _Float16* Qf  = (_Float16*)(ws + 256);                  // [BH][T][MP] f16
  size_t featElems = (size_t)BH * T_ * MP;
  _Float16* Kf  = Qf + featElems;                         // [BH][T][MP] f16
  _Float16* KfT = Kf + featElems;                         // [BH][MP][T] f16
  float* Spart  = (float*)(KfT + featElems);              // [BH][NC][80][MP] f32

  (void)hipMemsetAsync(kmaxEnc, 0, 256, stream);

  dim3 gFeat(T_ / CHUNK, BH);
  kmax_kernel<<<gFeat, 256, 0, stream>>>(key, proj, kmaxEnc);
  feat_kernel<<<gFeat, 256, 0, stream>>>(query, proj, kmaxEnc, Qf, (_Float16*)nullptr, 1);
  feat_kernel<<<gFeat, 256, 0, stream>>>(key, proj, kmaxEnc, Kf, KfT, 0);

  dim3 gChunk(NC, BH);
  chunkstate_kernel<<<gChunk, 320, 0, stream>>>(value, KfT, Spart);
  prefix_kernel<<<dim3(90, BH), 256, 0, stream>>>(Spart);
  attn_kernel<<<gChunk, 320, 0, stream>>>(value, Qf, Kf, KfT, Spart, out);
}
